// SelfConditioningResidualLayer_30554397343971
// MI455X (gfx1250) — matrix operand order repaired and checked by reading, not compiled, not run on hardware
//
#include <hip/hip_runtime.h>
#include <stdint.h>

// ---------------------------------------------------------------------------
// Types for CDNA5 WMMA (wave32): D(16x16 f32) = A(16x32 bf16) x B(32x16 bf16) + C
// ---------------------------------------------------------------------------
typedef float  v8f   __attribute__((ext_vector_type(8)));
typedef __bf16 v16bf __attribute__((ext_vector_type(16)));

union ABFrag { uint4 u4[2]; uint32_t u[8]; v16bf v; };

__device__ inline uint16_t f2bf(float f) {
  uint32_t u = __float_as_uint(f);
  uint32_t r = u + 0x7FFFu + ((u >> 16) & 1u);   // round-to-nearest-even
  return (uint16_t)(r >> 16);
}
__device__ inline uint32_t pack2(float a, float b) {
  return (uint32_t)f2bf(a) | ((uint32_t)f2bf(b) << 16);
}
__device__ inline float silu(float x) { return x / (1.0f + __expf(-x)); }

#define MU_STEP (20.0f / 15.0f)   // linspace(0,20,16) step
#define INV_SIG 0.8f              // 1 / (20/16)

// ---------------------------------------------------------------------------
// Workspace layout (uint32 units): bf16 pair-packed B matrices, Bp[n][kpair]
// ---------------------------------------------------------------------------
#define OFF_NW1 0            // 256 cols x 144 kpairs (K=288)
#define OFF_NW2 36864        // 256 cols x 128 kpairs (K=256)
#define OFF_EW1 69632        // 64 cols  x 48  kpairs (K=85 padded to 96)
#define OFF_EW2 72704        // 64 cols  x 32  kpairs (K=64)
#define WS_TOTAL 74752

__device__ inline uint32_t pack_w(const float* __restrict__ W, int K, int NC, int n, int kp) {
  int k0 = 2 * kp, k1 = k0 + 1;
  float a = (k0 < K) ? W[(size_t)k0 * NC + n] : 0.0f;
  float b = (k1 < K) ? W[(size_t)k1 * NC + n] : 0.0f;
  return pack2(a, b);
}

__global__ void pack_weights(const float* __restrict__ nW1, const float* __restrict__ nW2,
                             const float* __restrict__ eW1, const float* __restrict__ eW2,
                             uint32_t* __restrict__ ws) {
  int i = blockIdx.x * 256 + threadIdx.x;
  if (i < 36864) {
    int n = i / 144, kp = i % 144;
    ws[OFF_NW1 + i] = pack_w(nW1, 288, 256, n, kp);
  } else if (i < 69632) {
    int j = i - 36864; int n = j / 128, kp = j % 128;
    ws[OFF_NW2 + j] = pack_w(nW2, 256, 256, n, kp);
  } else if (i < 72704) {
    int j = i - 69632; int n = j / 48, kp = j % 48;
    ws[OFF_EW1 + j] = pack_w(eW1, 85, 64, n, kp);
  } else if (i < WS_TOTAL) {
    int j = i - 72704; int n = j / 32, kp = j % 32;
    ws[OFF_EW2 + j] = pack_w(eW2, 64, 64, n, kp);
  }
}

// ---------------------------------------------------------------------------
// Node branch: 32 rows/block, 4 waves; each wave: 2 row-groups x 64 cols.
// Every B fragment feeds 2 WMMAs (32 FLOP / L2 byte on the weight stream).
// MLP: 288 -> 256 -> 256 (SiLU), residual with h.
// ---------------------------------------------------------------------------
__global__ __launch_bounds__(128) void node_kernel(
    const float* __restrict__ h, const float* __restrict__ coord,
    const float* __restrict__ pos_head, const float* __restrict__ class_head,
    const uint32_t* __restrict__ W1p, const float* __restrict__ b1,
    const uint32_t* __restrict__ W2p, const float* __restrict__ b2,
    float* __restrict__ out, int N) {
  __shared__ uint32_t As[32 * 144];   // 32 rows x 288 bf16 (pair-packed)
  __shared__ uint32_t Hs[32 * 136];   // hidden 32 x 256 bf16, padded stride 272 halves

  const int tid  = threadIdx.x;
  const int lane = tid & 31;
  const int wave = tid >> 5;
  const int g    = lane >> 4;      // lane group (K-half select)
  const int nn   = lane & 15;      // M within tile for A, N for B/C
  const int rowBase = blockIdx.x * 32;

  // ---- stage h (cols 0..255): 4 threads per row, 64 cols each ----
  {
    int r = tid >> 2, p = tid & 3;
    int node = min(rowBase + r, N - 1);
    const float4* hr = (const float4*)(h + (size_t)node * 256 + p * 64);
    uint32_t* dst = &As[r * 144 + p * 32];
#pragma unroll
    for (int j = 0; j < 16; ++j) {
      float4 x = hr[j];
      dst[2 * j]     = pack2(x.x, x.y);
      dst[2 * j + 1] = pack2(x.z, x.w);
    }
  }
  // ---- per-row extras: softmax(16) + rbf(16) -> cols 256..287 ----
  if (tid < 32) {
    int node = min(rowBase + tid, N - 1);
    const float* ch = class_head + (size_t)node * 16;
    float e[32];
    float mx = ch[0];
#pragma unroll
    for (int i = 1; i < 16; ++i) mx = fmaxf(mx, ch[i]);
    float s = 0.0f;
#pragma unroll
    for (int i = 0; i < 16; ++i) { float v = __expf(ch[i] - mx); e[i] = v; s += v; }
    float inv = 1.0f / s;
#pragma unroll
    for (int i = 0; i < 16; ++i) e[i] *= inv;
    float dx = coord[node * 3 + 0] - pos_head[node * 3 + 0];
    float dy = coord[node * 3 + 1] - pos_head[node * 3 + 1];
    float dz = coord[node * 3 + 2] - pos_head[node * 3 + 2];
    float d = sqrtf(fmaxf(dx * dx + dy * dy + dz * dz, 1e-8f));
#pragma unroll
    for (int i = 0; i < 16; ++i) { float z = (d - MU_STEP * i) * INV_SIG; e[16 + i] = __expf(-z * z); }
    uint32_t* dst = &As[tid * 144 + 128];
#pragma unroll
    for (int j = 0; j < 16; ++j) dst[j] = pack2(e[2 * j], e[2 * j + 1]);
  }
  __syncthreads();

  const int colW = wave * 64;
  const v8f vzero = {0.f, 0.f, 0.f, 0.f, 0.f, 0.f, 0.f, 0.f};

  // ---- GEMM1: (32x288) x (288x256), this wave: 2 row-groups x 64 cols ----
  v8f c1[2][4];
#pragma unroll
  for (int rg = 0; rg < 2; ++rg)
#pragma unroll
    for (int t = 0; t < 4; ++t) c1[rg][t] = vzero;
  for (int kc = 0; kc < 288; kc += 32) {
    ABFrag a[2];
#pragma unroll
    for (int rg = 0; rg < 2; ++rg) {
      const uint32_t* ap = &As[(rg * 16 + nn) * 144 + (kc >> 1) + g * 8];
      a[rg].u4[0] = *(const uint4*)ap;
      a[rg].u4[1] = *(const uint4*)(ap + 4);
    }
#pragma unroll
    for (int t = 0; t < 4; ++t) {
      int n = colW + t * 16 + nn;
      const uint32_t* bp = &W1p[n * 144 + (kc >> 1) + g * 8];
      ABFrag b;
      b.u4[0] = *(const uint4*)bp;
      b.u4[1] = *(const uint4*)(bp + 4);
      c1[0][t] = __builtin_amdgcn_wmma_f32_16x16x32_bf16(false, a[0].v, false, b.v,
                                                         (short)0, c1[0][t], false, false);
      c1[1][t] = __builtin_amdgcn_wmma_f32_16x16x32_bf16(false, a[1].v, false, b.v,
                                                         (short)0, c1[1][t], false, false);
    }
  }

  // ---- SiLU + store hidden to LDS as bf16 ----
  {
    uint16_t* Hb = (uint16_t*)Hs;
#pragma unroll
    for (int t = 0; t < 4; ++t) {
      int n = colW + t * 16 + nn;
      float bias = b1[n];
#pragma unroll
      for (int rg = 0; rg < 2; ++rg)
#pragma unroll
        for (int v = 0; v < 8; ++v) {
          int m = rg * 16 + v + 8 * g;
          Hb[m * 272 + n] = f2bf(silu(c1[rg][t][v] + bias));
        }
    }
  }
  __syncthreads();

  // ---- GEMM2: (32x256) x (256x256) ----
  v8f c2[2][4];
#pragma unroll
  for (int rg = 0; rg < 2; ++rg)
#pragma unroll
    for (int t = 0; t < 4; ++t) c2[rg][t] = vzero;
  for (int kc = 0; kc < 256; kc += 32) {
    ABFrag a[2];
#pragma unroll
    for (int rg = 0; rg < 2; ++rg) {
      const uint32_t* ap = &Hs[(rg * 16 + nn) * 136 + (kc >> 1) + g * 8];
      a[rg].u4[0] = *(const uint4*)ap;
      a[rg].u4[1] = *(const uint4*)(ap + 4);
    }
#pragma unroll
    for (int t = 0; t < 4; ++t) {
      int n = colW + t * 16 + nn;
      const uint32_t* bp = &W2p[n * 128 + (kc >> 1) + g * 8];
      ABFrag b;
      b.u4[0] = *(const uint4*)bp;
      b.u4[1] = *(const uint4*)(bp + 4);
      c2[0][t] = __builtin_amdgcn_wmma_f32_16x16x32_bf16(false, a[0].v, false, b.v,
                                                         (short)0, c2[0][t], false, false);
      c2[1][t] = __builtin_amdgcn_wmma_f32_16x16x32_bf16(false, a[1].v, false, b.v,
                                                         (short)0, c2[1][t], false, false);
    }
  }

  // ---- SiLU + residual + store ----
#pragma unroll
  for (int t = 0; t < 4; ++t) {
    int n = colW + t * 16 + nn;
    float bias = b2[n];
#pragma unroll
    for (int rg = 0; rg < 2; ++rg)
#pragma unroll
      for (int v = 0; v < 8; ++v) {
        int node = rowBase + rg * 16 + v + 8 * g;
        if (node < N) {
          float x = silu(c2[rg][t][v] + bias);
          out[(size_t)node * 256 + n] = h[(size_t)node * 256 + n] + x;
        }
      }
  }
}

// ---------------------------------------------------------------------------
// Edge branch: 128 edges/block; 4 waves, each: 2 row-groups (32 edges) x 64 cols.
// MLP: 85(pad 96) -> 64 -> 64 (SiLU), residual with edge_attr.
// ---------------------------------------------------------------------------
__global__ __launch_bounds__(128) void edge_kernel(
    const float* __restrict__ coord, const int* __restrict__ eidx,
    const float* __restrict__ edge_attr, const float* __restrict__ pos_head,
    const float* __restrict__ etype,
    const uint32_t* __restrict__ W1p, const float* __restrict__ b1,
    const uint32_t* __restrict__ W2p, const float* __restrict__ b2,
    float* __restrict__ out, long long E) {
  __shared__ uint32_t Es[128 * 48];   // 128 rows x 96 bf16 (pair-packed)
  __shared__ uint32_t Hs[128 * 36];   // hidden 128 x 64 bf16, padded stride 72 halves

  const int tid  = threadIdx.x;
  const int lane = tid & 31;
  const int wave = tid >> 5;
  const int g    = lane >> 4;
  const int nn   = lane & 15;
  const long long eBase = (long long)blockIdx.x * 128;

  // ---- stage one full input row per thread ----
  {
    long long e = eBase + tid; if (e >= E) e = E - 1;
    // edge_attr cols 0..63
    const float4* ar = (const float4*)(edge_attr + e * 64);
    uint32_t* dst = &Es[tid * 48];
#pragma unroll
    for (int j = 0; j < 16; ++j) {
      float4 x = ar[j];
      dst[2 * j]     = pack2(x.x, x.y);
      dst[2 * j + 1] = pack2(x.z, x.w);
    }
    // extras: softmax(5) + rbf-diff(16) + zero pad(11) -> cols 64..95
    int ri = eidx[e], ci = eidx[E + e];
    float ex[32];
    const float* et = etype + e * 5;
    float mx = et[0];
#pragma unroll
    for (int i = 1; i < 5; ++i) mx = fmaxf(mx, et[i]);
    float s = 0.0f;
#pragma unroll
    for (int i = 0; i < 5; ++i) { float v = __expf(et[i] - mx); ex[i] = v; s += v; }
    float inv = 1.0f / s;
#pragma unroll
    for (int i = 0; i < 5; ++i) ex[i] *= inv;
    float p0x = pos_head[ri * 3 + 0] - pos_head[ci * 3 + 0];
    float p0y = pos_head[ri * 3 + 1] - pos_head[ci * 3 + 1];
    float p0z = pos_head[ri * 3 + 2] - pos_head[ci * 3 + 2];
    float c0x = coord[ri * 3 + 0] - coord[ci * 3 + 0];
    float c0y = coord[ri * 3 + 1] - coord[ci * 3 + 1];
    float c0z = coord[ri * 3 + 2] - coord[ci * 3 + 2];
    float d1 = sqrtf(fmaxf(p0x * p0x + p0y * p0y + p0z * p0z, 1e-8f)) + 1e-8f;
    float d0 = sqrtf(fmaxf(c0x * c0x + c0y * c0y + c0z * c0z, 1e-8f)) + 1e-8f;
#pragma unroll
    for (int i = 0; i < 16; ++i) {
      float z1 = (d1 - MU_STEP * i) * INV_SIG;
      float z0 = (d0 - MU_STEP * i) * INV_SIG;
      ex[5 + i] = __expf(-z1 * z1) - __expf(-z0 * z0);
    }
#pragma unroll
    for (int i = 21; i < 32; ++i) ex[i] = 0.0f;
    uint32_t* dste = &Es[tid * 48 + 32];
#pragma unroll
    for (int j = 0; j < 16; ++j) dste[j] = pack2(ex[2 * j], ex[2 * j + 1]);
  }
  __syncthreads();

  const v8f vzero = {0.f, 0.f, 0.f, 0.f, 0.f, 0.f, 0.f, 0.f};
  const int rowOff = wave * 32;   // this wave's 32-edge slab (2 row groups)

  // ---- GEMM1: (32x96) x (96x64) ----
  v8f c1[2][4];
#pragma unroll
  for (int rg = 0; rg < 2; ++rg)
#pragma unroll
    for (int t = 0; t < 4; ++t) c1[rg][t] = vzero;
  for (int kc = 0; kc < 96; kc += 32) {
    ABFrag a[2];
#pragma unroll
    for (int rg = 0; rg < 2; ++rg) {
      const uint32_t* ap = &Es[(rowOff + rg * 16 + nn) * 48 + (kc >> 1) + g * 8];
      a[rg].u4[0] = *(const uint4*)ap;
      a[rg].u4[1] = *(const uint4*)(ap + 4);
    }
#pragma unroll
    for (int t = 0; t < 4; ++t) {
      int n = t * 16 + nn;
      const uint32_t* bp = &W1p[n * 48 + (kc >> 1) + g * 8];
      ABFrag b;
      b.u4[0] = *(const uint4*)bp;
      b.u4[1] = *(const uint4*)(bp + 4);
      c1[0][t] = __builtin_amdgcn_wmma_f32_16x16x32_bf16(false, a[0].v, false, b.v,
                                                         (short)0, c1[0][t], false, false);
      c1[1][t] = __builtin_amdgcn_wmma_f32_16x16x32_bf16(false, a[1].v, false, b.v,
                                                         (short)0, c1[1][t], false, false);
    }
  }

  // ---- SiLU + hidden to LDS ----
  {
    uint16_t* Hb = (uint16_t*)Hs;
#pragma unroll
    for (int t = 0; t < 4; ++t) {
      int n = t * 16 + nn;
      float bias = b1[n];
#pragma unroll
      for (int rg = 0; rg < 2; ++rg)
#pragma unroll
        for (int v = 0; v < 8; ++v) {
          int m = rowOff + rg * 16 + v + 8 * g;
          Hb[m * 72 + n] = f2bf(silu(c1[rg][t][v] + bias));
        }
    }
  }
  __syncthreads();

  // ---- GEMM2: (32x64) x (64x64) ----
  v8f c2[2][4];
#pragma unroll
  for (int rg = 0; rg < 2; ++rg)
#pragma unroll
    for (int t = 0; t < 4; ++t) c2[rg][t] = vzero;
  for (int kc = 0; kc < 64; kc += 32) {
    ABFrag a[2];
#pragma unroll
    for (int rg = 0; rg < 2; ++rg) {
      const uint32_t* ap = &Hs[(rowOff + rg * 16 + nn) * 36 + (kc >> 1) + g * 8];
      a[rg].u4[0] = *(const uint4*)ap;
      a[rg].u4[1] = *(const uint4*)(ap + 4);
    }
#pragma unroll
    for (int t = 0; t < 4; ++t) {
      int n = t * 16 + nn;
      const uint32_t* bp = &W2p[n * 32 + (kc >> 1) + g * 8];
      ABFrag b;
      b.u4[0] = *(const uint4*)bp;
      b.u4[1] = *(const uint4*)(bp + 4);
      c2[0][t] = __builtin_amdgcn_wmma_f32_16x16x32_bf16(false, a[0].v, false, b.v,
                                                         (short)0, c2[0][t], false, false);
      c2[1][t] = __builtin_amdgcn_wmma_f32_16x16x32_bf16(false, a[1].v, false, b.v,
                                                         (short)0, c2[1][t], false, false);
    }
  }

  // ---- SiLU + residual + store ----
#pragma unroll
  for (int t = 0; t < 4; ++t) {
    int n = t * 16 + nn;
    float bias = b2[n];
#pragma unroll
    for (int rg = 0; rg < 2; ++rg)
#pragma unroll
      for (int v = 0; v < 8; ++v) {
        long long e = eBase + rowOff + rg * 16 + v + 8 * g;
        if (e < E) {
          float x = silu(c2[rg][t][v] + bias);
          out[e * 64 + n] = edge_attr[e * 64 + n] + x;
        }
      }
  }
}

// ---------------------------------------------------------------------------
__global__ void copy_coord(const float* __restrict__ coord, float* __restrict__ out, int n) {
  int i = blockIdx.x * 256 + threadIdx.x;
  if (i < n) out[i] = coord[i];
}

// ---------------------------------------------------------------------------
extern "C" void kernel_launch(void* const* d_in, const int* in_sizes, int n_in,
                              void* d_out, int out_size, void* d_ws, size_t ws_size,
                              hipStream_t stream) {
  (void)n_in; (void)out_size; (void)ws_size;
  const float* h          = (const float*)d_in[0];
  const float* coord      = (const float*)d_in[1];
  const int*   edge_index = (const int*)d_in[2];
  const float* edge_attr  = (const float*)d_in[3];
  const float* pos_head   = (const float*)d_in[4];
  const float* class_head = (const float*)d_in[5];
  const float* etype      = (const float*)d_in[6];
  const float* nW1 = (const float*)d_in[7];
  const float* nb1 = (const float*)d_in[8];
  const float* nW2 = (const float*)d_in[9];
  const float* nb2 = (const float*)d_in[10];
  const float* eW1 = (const float*)d_in[11];
  const float* eb1 = (const float*)d_in[12];
  const float* eW2 = (const float*)d_in[13];
  const float* eb2 = (const float*)d_in[14];

  const int N = in_sizes[0] / 256;
  const long long E = (long long)in_sizes[3] / 64;

  float* out = (float*)d_out;
  uint32_t* ws = (uint32_t*)d_ws;

  pack_weights<<<(WS_TOTAL + 255) / 256, 256, 0, stream>>>(nW1, nW2, eW1, eW2, ws);

  node_kernel<<<(N + 31) / 32, 128, 0, stream>>>(
      h, coord, pos_head, class_head,
      ws + OFF_NW1, nb1, ws + OFF_NW2, nb2, out, N);

  float* edgeOut = out + (size_t)N * 256 + (size_t)N * 3;
  edge_kernel<<<(int)((E + 127) / 128), 128, 0, stream>>>(
      coord, edge_index, edge_attr, pos_head, etype,
      ws + OFF_EW1, eb1, ws + OFF_EW2, eb2, edgeOut, E);

  copy_coord<<<(N * 3 + 255) / 256, 256, 0, stream>>>(coord, out + (size_t)N * 256, N * 3);
}
